// WeightedChildSumTreeLSTMEndtoEnd_32238024524220
// MI455X (gfx1250) — compile-verified, matrix operand order checked by reading, not run on hardware
//
#include <hip/hip_runtime.h>
#include <hip/hip_bf16.h>

// ---------------- types ----------------
typedef __bf16 bf16;
typedef __attribute__((ext_vector_type(16))) __bf16 v16bf;
typedef __attribute__((ext_vector_type(8)))  __bf16 v8bf;
typedef __attribute__((ext_vector_type(4)))  __bf16 v4bf;
typedef __attribute__((ext_vector_type(8)))  float  v8f;
typedef __attribute__((ext_vector_type(4)))  unsigned int v4u;
typedef __attribute__((ext_vector_type(8)))  unsigned int v8u;

// ---------------- problem constants ----------------
#define NN      4681      // total nodes
#define BSZ     64        // batch
#define MEMD    256
#define INDIM   256
#define NODE_EL (BSZ*MEMD)          // 16384 activations per node
#define LDSB    264                 // bf16 LDS row stride (256 + 8 pad)
#define LDSF    260                 // f32  LDS row stride (256 + 4 pad)

#define SM_XBF   (BSZ*LDSB*2)                 // 33792 B
#define SM_HBF   (BSZ*LDSB*2)                 // 33792 B (x2 buffers)
#define SM_HSUM  (BSZ*LDSF*4)                 // 66560 B
#define SM_FX    (BSZ*LDSF*4)
#define SM_ACC   (BSZ*LDSF*4)
#define SMEM_LEAF (SM_XBF)
#define SMEM_INT  (SM_XBF + 2*SM_HBF + SM_HSUM + SM_FX + SM_ACC) // 301056 B

// ---------------- fast math ----------------
__device__ __forceinline__ float fast_rcp(float x) {
#if __has_builtin(__builtin_amdgcn_rcpf)
  return __builtin_amdgcn_rcpf(x);
#else
  return 1.0f / x;
#endif
}
__device__ __forceinline__ float fast_tanh(float x) {
#if __has_builtin(__builtin_amdgcn_tanhf)
  return __builtin_amdgcn_tanhf(x);
#else
  return tanhf(x);
#endif
}
__device__ __forceinline__ float sigm(float x) {
  return fast_rcp(1.0f + __expf(-x));
}

// ---------------- WMMA helpers ----------------
__device__ __forceinline__ v8f wmma_bf16(v16bf a, v16bf b, v8f c) {
  return __builtin_amdgcn_wmma_f32_16x16x32_bf16(
      false, a, false, b, (short)0, c, false, false);
}

// 16x32 bf16 fragment load (A layout; also B when matrix is row-major with
// rows = N, cols = K, i.e. the weight matrices as stored).
__device__ __forceinline__ v16bf load_frag(const bf16* __restrict__ base,
                                           int row0, int stride, int k0) {
  const int lane = threadIdx.x & 31;
  const bf16* p = base + (size_t)(row0 + (lane & 15)) * stride
                       + k0 + ((lane >> 4) << 3);
  v8bf lo = *(const v8bf*)(p);
  v8bf hi = *(const v8bf*)(p + 16);
  return __builtin_shufflevector(lo, hi, 0,1,2,3,4,5,6,7,8,9,10,11,12,13,14,15);
}

// ---------------- TDM: DMA 8 rows x 256 bf16 into padded LDS rows ----------
// D# per CDNA5 ISA ch.8: group0 {count, lds_addr, global_addr, type=2};
// group1 {data_size=2B, pad_enable, pad_interval=6 (128 DW), pad_amount=3
// (4 DW) -> LDS row stride 264 bf16, tensor_dim0=256, tensor_dim1=8,
// tile_dim0=256, tile_dim1=8, dim0_stride=256}. Groups 2/3 zero (2-D tile).
__device__ __forceinline__ void tdm_load_8rows(unsigned lds_off,
                                               const bf16* __restrict__ gsrc) {
  unsigned long long ga = (unsigned long long)(size_t)gsrc;
  v4u g0 = { 1u,                                   // count = 1 valid D#
             lds_off,                              // LDS byte address
             (unsigned)ga,
             ((unsigned)(ga >> 32) & 0x01FFFFFFu) | 0x80000000u }; // type=2
  v8u g1 = { (1u << 16) | (1u << 20) | (6u << 22) | (3u << 25),
             (256u << 16),                         // tensor_dim0 = 256
             (8u   << 16),                         // tensor_dim1 = 8
             (256u << 16),                         // tile_dim0 = 256
             8u,                                   // tile_dim1 = 8
             256u,                                 // tensor_dim0_stride = 256
             0u, 0u };
  v4u g2 = {0u, 0u, 0u, 0u};
  v4u g3 = {0u, 0u, 0u, 0u};
  asm volatile("tensor_load_to_lds %0, %1, %2, %3"
               :: "s"(g0), "s"(g1), "s"(g2), "s"(g3)
               : "memory");
}

// ---------------- weight f32 -> bf16 ----------------
__global__ void cvt_f32_bf16(const float* __restrict__ src, bf16* __restrict__ dst, int n) {
  int i = blockIdx.x * 256 + threadIdx.x;
  if (i < n) dst[i] = (bf16)src[i];
}

// ---------------- leaf level ----------------
__global__ __launch_bounds__(256) void tree_leaf(
    const float* __restrict__ inputs,
    const bf16*  __restrict__ Wioux,
    const float* __restrict__ bioux, const float* __restrict__ biouh,
    bf16* __restrict__ c_all, bf16* __restrict__ h_all, int start)
{
  extern __shared__ char smem[];
  bf16* sx = (bf16*)smem;

  const int gnode = start + blockIdx.x;
  const int tid  = threadIdx.x;
  const int wave = tid >> 5;
  const int lane = tid & 31;
  const int half = lane >> 4;
  const int ncol = lane & 15;

  const float* xg = inputs + (size_t)gnode * NODE_EL;
  #pragma unroll
  for (int i = 0; i < 16; ++i) {
    int f4 = tid + i * 256;
    int e = f4 * 4; int row = e >> 8; int col = e & 255;
    float4 v = ((const float4*)xg)[f4];
    v4bf b; b[0]=(bf16)v.x; b[1]=(bf16)v.y; b[2]=(bf16)v.z; b[3]=(bf16)v.w;
    *(v4bf*)&sx[row * LDSB + col] = b;
  }
  __syncthreads();

  bf16* cdst = c_all + (size_t)gnode * NODE_EL;
  bf16* hdst = h_all + (size_t)gnode * NODE_EL;

  const int n0 = wave * 32;
  for (int hn = 0; hn < 2; ++hn) {
    const int nc = n0 + hn * 16;
    for (int m0 = 0; m0 < BSZ; m0 += 16) {
      v8f ai = {0,0,0,0,0,0,0,0}, ao = ai, au = ai;
      #pragma unroll
      for (int k = 0; k < INDIM; k += 32) {
        v16bf a = load_frag(sx, m0, LDSB, k);
        ai = wmma_bf16(a, load_frag(Wioux, nc,        INDIM, k), ai);
        ao = wmma_bf16(a, load_frag(Wioux, nc + 256,  INDIM, k), ao);
        au = wmma_bf16(a, load_frag(Wioux, nc + 512,  INDIM, k), au);
      }
      #pragma unroll
      for (int r = 0; r < 8; ++r) {
        int m = m0 + r + half * 8; int n = nc + ncol;
        float iv = ai[r] + bioux[n]       + biouh[n];
        float ov = ao[r] + bioux[n + 256] + biouh[n + 256];
        float uv = au[r] + bioux[n + 512] + biouh[n + 512];
        float cc = sigm(iv) * fast_tanh(uv);
        float hh = sigm(ov) * fast_tanh(cc);
        cdst[m * MEMD + n] = (bf16)cc;
        hdst[m * MEMD + n] = (bf16)hh;
      }
    }
  }
}

// ---------------- internal level ----------------
__global__ __launch_bounds__(256) void tree_internal(
    const float* __restrict__ inputs, const float* __restrict__ prob,
    const bf16*  __restrict__ Wioux, const float* __restrict__ bioux,
    const bf16*  __restrict__ Wiouh, const float* __restrict__ biouh,
    const bf16*  __restrict__ Wfx,   const float* __restrict__ bfx,
    const bf16*  __restrict__ Wfh,   const float* __restrict__ bfh,
    bf16* __restrict__ c_all, bf16* __restrict__ h_all,
    int start, float* __restrict__ out)
{
  extern __shared__ char smem[];
  bf16*  sx    = (bf16*)(smem);
  bf16*  shA   = (bf16*)(smem + SM_XBF);
  bf16*  shB   = (bf16*)(smem + SM_XBF + SM_HBF);
  float* shsum = (float*)(smem + SM_XBF + 2*SM_HBF);
  float* sfx   = (float*)(smem + SM_XBF + 2*SM_HBF + SM_HSUM);
  float* sacc  = (float*)(smem + SM_XBF + 2*SM_HBF + SM_HSUM + SM_FX);

  const int gnode = start + blockIdx.x;
  const int tid  = threadIdx.x;
  const int wave = tid >> 5;
  const int lane = tid & 31;
  const int half = lane >> 4;
  const int ncol = lane & 15;
  const int n0   = wave * 32;

  // scalar (wave-uniform) values for TDM descriptors
  const int wv = __builtin_amdgcn_readfirstlane(wave);
  const unsigned ldsA = (unsigned)(size_t)(void*)shA + (unsigned)(wv * 8 * LDSB * 2);
  const unsigned ldsB = (unsigned)(size_t)(void*)shB + (unsigned)(wv * 8 * LDSB * 2);

  // ---- kick off DMA of child 0's h slice (8 rows per wave) ----
  {
    const int child0 = 8 * gnode + 1;
    tdm_load_8rows(ldsA, h_all + (size_t)child0 * NODE_EL + wv * 8 * MEMD);
  }

  // ---- stage x -> bf16 LDS (overlaps with TDM) ----
  const float* xg = inputs + (size_t)gnode * NODE_EL;
  #pragma unroll
  for (int i = 0; i < 16; ++i) {
    int f4 = tid + i * 256;
    int e = f4 * 4; int row = e >> 8; int col = e & 255;
    float4 v = ((const float4*)xg)[f4];
    v4bf b; b[0]=(bf16)v.x; b[1]=(bf16)v.y; b[2]=(bf16)v.z; b[3]=(bf16)v.w;
    *(v4bf*)&sx[row * LDSB + col] = b;
  }
  __syncthreads();

  // ---- fx = x @ Wfx^T + bfx -> sfx (also overlaps with TDM) ----
  for (int hn = 0; hn < 2; ++hn) {
    const int nc = n0 + hn * 16;
    for (int m0 = 0; m0 < BSZ; m0 += 16) {
      v8f acc = {0,0,0,0,0,0,0,0};
      #pragma unroll
      for (int k = 0; k < INDIM; k += 32)
        acc = wmma_bf16(load_frag(sx, m0, LDSB, k),
                        load_frag(Wfx, nc, INDIM, k), acc);
      #pragma unroll
      for (int r = 0; r < 8; ++r) {
        int m = m0 + r + half * 8; int n = nc + ncol;
        sfx[m * LDSF + n] = acc[r] + bfx[n];
      }
    }
  }

  // ---- children loop (double-buffered TDM pipeline) ----
  for (int k = 0; k < 8; ++k) {
    const int child = 8 * gnode + 1 + k;
    const float pw = prob[(size_t)gnode * NN + child];
    const bf16* cg = c_all + (size_t)child * NODE_EL;
    bf16* cur = (k & 1) ? shB : shA;

    __builtin_amdgcn_s_wait_tensorcnt(0);   // this wave's slice of child k landed
    __syncthreads();                        // all slices landed; prev iter done

    if (k < 7) {                            // DMA child k+1 behind this compute
      const unsigned dst = (k & 1) ? ldsA : ldsB;
      tdm_load_8rows(dst, h_all + (size_t)(child + 1) * NODE_EL + wv * 8 * MEMD);
    }

    // h_sum += pw * h_k  (f32 accumulation from raw bf16 in LDS)
    #pragma unroll
    for (int i = 0; i < 8; ++i) {
      int f8 = tid + i * 256;               // 0..2047 (8-elem chunks)
      int row = f8 >> 5; int col = (f8 & 31) * 8;
      v8bf hv = *(const v8bf*)&cur[row * LDSB + col];
      float* ss = &shsum[row * LDSF + col];
      #pragma unroll
      for (int j = 0; j < 8; ++j) {
        float wh = pw * (float)hv[j];
        ss[j] = (k == 0) ? wh : (ss[j] + wh);
      }
    }

    // f_k = sigmoid(pw*(h_k@Wfh^T) + bfh + fx); c_partial += f_k * pw * c_k
    for (int hn = 0; hn < 2; ++hn) {
      const int nc = n0 + hn * 16;
      for (int m0 = 0; m0 < BSZ; m0 += 16) {
        v8f acc = {0,0,0,0,0,0,0,0};
        #pragma unroll
        for (int kk = 0; kk < MEMD; kk += 32)
          acc = wmma_bf16(load_frag(cur, m0, LDSB, kk),
                          load_frag(Wfh, nc, MEMD, kk), acc);
        #pragma unroll
        for (int r = 0; r < 8; ++r) {
          int m = m0 + r + half * 8; int n = nc + ncol;
          float f  = sigm(pw * acc[r] + bfh[n] + sfx[m * LDSF + n]);
          float wc = pw * (float)cg[m * MEMD + n];
          float prev = (k == 0) ? 0.0f : sacc[m * LDSF + n];
          sacc[m * LDSF + n] = prev + f * wc;
        }
      }
    }
  }
  __syncthreads();

  // ---- h_sum (f32) -> bf16 into shA ----
  #pragma unroll
  for (int i = 0; i < 64; ++i) {
    int e = tid + i * 256;
    int row = e >> 8; int col = e & 255;
    shA[row * LDSB + col] = (bf16)shsum[row * LDSF + col];
  }
  __syncthreads();

  // ---- iou = x@Wioux^T + h_sum@Wiouh^T + biases; gates; write c,h ----
  bf16* cdst = c_all + (size_t)gnode * NODE_EL;
  bf16* hdst = h_all + (size_t)gnode * NODE_EL;

  for (int hn = 0; hn < 2; ++hn) {
    const int nc = n0 + hn * 16;
    for (int m0 = 0; m0 < BSZ; m0 += 16) {
      v8f ai = {0,0,0,0,0,0,0,0}, ao = ai, au = ai;
      #pragma unroll
      for (int kk = 0; kk < INDIM; kk += 32) {
        v16bf a = load_frag(sx, m0, LDSB, kk);
        ai = wmma_bf16(a, load_frag(Wioux, nc,       INDIM, kk), ai);
        ao = wmma_bf16(a, load_frag(Wioux, nc + 256, INDIM, kk), ao);
        au = wmma_bf16(a, load_frag(Wioux, nc + 512, INDIM, kk), au);
      }
      #pragma unroll
      for (int kk = 0; kk < MEMD; kk += 32) {
        v16bf a = load_frag(shA, m0, LDSB, kk);
        ai = wmma_bf16(a, load_frag(Wiouh, nc,       MEMD, kk), ai);
        ao = wmma_bf16(a, load_frag(Wiouh, nc + 256, MEMD, kk), ao);
        au = wmma_bf16(a, load_frag(Wiouh, nc + 512, MEMD, kk), au);
      }
      #pragma unroll
      for (int r = 0; r < 8; ++r) {
        int m = m0 + r + half * 8; int n = nc + ncol;
        float iv = ai[r] + bioux[n]       + biouh[n];
        float ov = ao[r] + bioux[n + 256] + biouh[n + 256];
        float uv = au[r] + bioux[n + 512] + biouh[n + 512];
        float cc = sigm(iv) * fast_tanh(uv) + sacc[m * LDSF + n];
        float hh = sigm(ov) * fast_tanh(cc);
        cdst[m * MEMD + n] = (bf16)cc;
        hdst[m * MEMD + n] = (bf16)hh;
        if (out) {                          // only the root launch passes out
          out[m * MEMD + n]           = cc;
          out[NODE_EL + m * MEMD + n] = hh;
        }
      }
    }
  }
}

// ---------------- host launcher ----------------
extern "C" void kernel_launch(void* const* d_in, const int* in_sizes, int n_in,
                              void* d_out, int out_size, void* d_ws, size_t ws_size,
                              hipStream_t stream) {
  const float* inputs = (const float*)d_in[0];
  const float* prob   = (const float*)d_in[1];
  const float* Wioux  = (const float*)d_in[2];
  const float* bioux  = (const float*)d_in[3];
  const float* Wiouh  = (const float*)d_in[4];
  const float* biouh  = (const float*)d_in[5];
  const float* Wfx    = (const float*)d_in[6];
  const float* bfx    = (const float*)d_in[7];
  const float* Wfh    = (const float*)d_in[8];
  const float* bfh    = (const float*)d_in[9];

  char* ws = (char*)d_ws;
  size_t off = 0;
  const size_t actB = (size_t)NN * NODE_EL * sizeof(bf16);   // 153.4 MB
  bf16* c_all = (bf16*)(ws + off); off += actB;
  bf16* h_all = (bf16*)(ws + off); off += actB;
  bf16* wioux_bf = (bf16*)(ws + off); off += (size_t)768 * 256 * 2;
  bf16* wiouh_bf = (bf16*)(ws + off); off += (size_t)768 * 256 * 2;
  bf16* wfx_bf   = (bf16*)(ws + off); off += (size_t)256 * 256 * 2;
  bf16* wfh_bf   = (bf16*)(ws + off); off += (size_t)256 * 256 * 2;

  cvt_f32_bf16<<<(768*256 + 255) / 256, 256, 0, stream>>>(Wioux, wioux_bf, 768*256);
  cvt_f32_bf16<<<(768*256 + 255) / 256, 256, 0, stream>>>(Wiouh, wiouh_bf, 768*256);
  cvt_f32_bf16<<<(256*256 + 255) / 256, 256, 0, stream>>>(Wfx,   wfx_bf,   256*256);
  cvt_f32_bf16<<<(256*256 + 255) / 256, 256, 0, stream>>>(Wfh,   wfh_bf,   256*256);

  // depth 4 (leaves): nodes 585..4680
  tree_leaf<<<4096, 256, SMEM_LEAF, stream>>>(inputs, wioux_bf, bioux, biouh,
                                              c_all, h_all, 585);

  // depths 3..0
  const int starts[4] = {73, 9, 1, 0};
  const int cnts[4]   = {512, 64, 8, 1};
  for (int d = 0; d < 4; ++d) {
    tree_internal<<<cnts[d], 256, SMEM_INT, stream>>>(
        inputs, prob,
        wioux_bf, bioux, wiouh_bf, biouh,
        wfx_bf, bfx, wfh_bf, bfh,
        c_all, h_all, starts[d],
        (d == 3) ? (float*)d_out : nullptr);
  }
}
